// lwm_mamba_31980326486864
// MI455X (gfx1250) — matrix operand order repaired and checked by reading, not compiled, not run on hardware
//
#include <hip/hip_runtime.h>
#include <hip/hip_bf16.h>

// ---------------------------------------------------------------------------
// CDNA5 (gfx1250) Mamba-stack forward.
//   - GEMMs: V_WMMA_F32_16X16X32_BF16, 128x128 macro-tile, 8 wmma/wave/K-step
//   - SSM scan: async global->LDS staging (ASYNCcnt) with double buffering
// ---------------------------------------------------------------------------

typedef __attribute__((ext_vector_type(16))) __bf16 v16bf;
typedef __attribute__((ext_vector_type(8)))  float  v8f;

union FragBF {
    v16bf        v;
    unsigned int u[8];
};

#define SEQ_L 512
#define DMODEL 256

__device__ __forceinline__ unsigned int pack_bf16(float a, float b) {
    unsigned int ua = __float_as_uint(a);
    unsigned int ub = __float_as_uint(b);
    ua = (ua + 0x7FFFu + ((ua >> 16) & 1u)) >> 16;   // round-to-nearest-even
    ub = (ub + 0x7FFFu + ((ub >> 16) & 1u)) >> 16;
    return (ua & 0xFFFFu) | (ub << 16);
}

__device__ __forceinline__ float silu_f(float x) {
    return x / (1.0f + expf(-x));
}
__device__ __forceinline__ float gelu_f(float x) {
    return 0.5f * x * (1.0f + erff(x * 0.70710678118654752f));
}
__device__ __forceinline__ float softplus_f(float x) {
    return (x > 20.0f) ? x : log1pf(expf(x));
}

// ---------------------------------------------------------------------------
// bf16 WMMA GEMM:  C[M,N] = epi(A[M,K] * B[K,N] + bias)
// Assumptions (hold for every call in this model): M % 128 == 0, K % 32 == 0,
// lda/ldb even. NGUARD only needed when N % 128 != 0 (the N=36 x-proj GEMM).
//   AREV / CREV: time-reverse the row index within each length-512 sequence
//   on the A read / C write (implements the backward SSM branch).
//   EPI: 0 = none, 1 = +bias, 2 = +bias then exact GELU
// Block: 256 threads = 8 waves (4 M x 2 N). Macro-tile 128x128, BK=32.
// Each wave: 2 A frags x 4 B frags -> 8 v_wmma per K-step.
// ---------------------------------------------------------------------------
template <int EPI, bool NGUARD, bool AREV, bool CREV>
__global__ __launch_bounds__(256)
void gemm_kernel(const float* __restrict__ A, const float* __restrict__ Bw,
                 const float* __restrict__ bias, float* __restrict__ C,
                 int M, int N, int K, int lda, int ldb, int ldc) {
    // LDS word [r][c] packs bf16 elements K = 2c, 2c+1 of row r
    __shared__ unsigned int As[128 * 16];   // A tile: 128(M) x 32(K)
    __shared__ unsigned int Bs[128 * 16];   // B tile transposed: 128(N) x 32(K)

    const int tid  = threadIdx.x;
    const int lane = tid & 31;
    const int wid  = tid >> 5;
    const int bm   = blockIdx.y * 128;
    const int bn   = blockIdx.x * 128;

    const int wm = (wid & 3) * 32;      // wave M offset (2 tiles)
    const int wn = (wid >> 2) * 64;     // wave N offset (4 tiles)

    const int lrow = lane & 15;
    const int hf   = lane >> 4;

    v8f acc[2][4];
    #pragma unroll
    for (int mi = 0; mi < 2; ++mi)
        #pragma unroll
        for (int ni = 0; ni < 4; ++ni)
            acc[mi][ni] = {};

    for (int kb = 0; kb < K; kb += 32) {
        // prefetch next A K-slab (global_prefetch_b8)
        if (kb + 32 < K) {
            int pr = bm + (tid & 127);
            __builtin_prefetch(A + (size_t)pr * lda + kb + 32, 0, 3);
        }
        // ---- stage A: 2048 words, 8 per thread, b64 loads, no guards ----
        #pragma unroll
        for (int i = 0; i < 8; ++i) {
            int w = tid + i * 256;
            int m = w >> 4;
            int c = w & 15;
            int gm = bm + m;
            int sr = AREV ? (((gm >> 9) << 9) + (511 - (gm & 511))) : gm;
            const float2 f2 = *(const float2*)(A + (size_t)sr * lda + kb + 2 * c);
            As[w] = pack_bf16(f2.x, f2.y);
        }
        // ---- stage B transposed (N-major, K contiguous) ----
        #pragma unroll
        for (int i = 0; i < 8; ++i) {
            int w = tid + i * 256;
            int n = w >> 4;
            int c = w & 15;
            int gn = bn + n;
            int k  = kb + 2 * c;
            float f0 = 0.0f, f1 = 0.0f;
            if (!NGUARD || gn < N) {
                f0 = Bw[(size_t)k       * ldb + gn];
                f1 = Bw[(size_t)(k + 1) * ldb + gn];
            }
            Bs[w] = pack_bf16(f0, f1);
        }
        __syncthreads();

        // A fragment (ISA 16-bit A 16x32 layout):
        //   lane half 0: K 0..7 then 16..23 ; half 1: K 8..15 then 24..31
        FragBF fa[2];
        #pragma unroll
        for (int mi = 0; mi < 2; ++mi) {
            int m  = wm + mi * 16 + lrow;
            int wb = hf * 4;
            #pragma unroll
            for (int j = 0; j < 4; ++j) {
                fa[mi].u[j]     = As[m * 16 + wb + j];
                fa[mi].u[4 + j] = As[m * 16 + wb + 8 + j];
            }
        }
        // B fragment (same K order as A: half 0: K 0..7 then 16..23 ; half 1: K 8..15 then 24..31)
        FragBF fb[4];
        #pragma unroll
        for (int ni = 0; ni < 4; ++ni) {
            int n  = wn + ni * 16 + lrow;
            int cb = hf * 4;
            #pragma unroll
            for (int j = 0; j < 4; ++j) {
                fb[ni].u[j]     = Bs[n * 16 + cb + j];
                fb[ni].u[4 + j] = Bs[n * 16 + cb + 8 + j];
            }
        }

        #pragma unroll
        for (int mi = 0; mi < 2; ++mi)
            #pragma unroll
            for (int ni = 0; ni < 4; ++ni)
                acc[mi][ni] = __builtin_amdgcn_wmma_f32_16x16x32_bf16(
                    false, fa[mi].v, false, fb[ni].v, (short)0,
                    acc[mi][ni], false, false);
        __syncthreads();
    }

    // epilogue: D tile layout -> lane N = lane&15, VGPR r -> M = r + 8*(lane>>4)
    #pragma unroll
    for (int mi = 0; mi < 2; ++mi) {
        #pragma unroll
        for (int ni = 0; ni < 4; ++ni) {
            int gn = bn + wn + ni * 16 + lrow;
            if (!NGUARD || gn < N) {
                float bv = (EPI >= 1) ? bias[gn] : 0.0f;
                #pragma unroll
                for (int r = 0; r < 8; ++r) {
                    int gm = bm + wm + mi * 16 + hf * 8 + r;
                    float v = acc[mi][ni][r] + bv;
                    if (EPI == 2) v = gelu_f(v);
                    int dr = CREV ? (((gm >> 9) << 9) + (511 - (gm & 511))) : gm;
                    C[(size_t)dr * ldc + gn] = v;
                }
            }
        }
    }
}

// ---------------------------------------------------------------------------
// Residual + LayerNorm over D=256: out = LN(in + res) * g + b
// ---------------------------------------------------------------------------
__global__ __launch_bounds__(256)
void ln_kernel(const float* __restrict__ in, const float* __restrict__ res,
               const float* __restrict__ g, const float* __restrict__ beta,
               float* __restrict__ out) {
    __shared__ float r[256];
    const int row = blockIdx.x;
    const int tid = threadIdx.x;
    const size_t idx = (size_t)row * DMODEL + tid;
    float v = in[idx] + (res ? res[idx] : 0.0f);
    r[tid] = v;
    __syncthreads();
    for (int o = 128; o > 0; o >>= 1) {
        if (tid < o) r[tid] += r[tid + o];
        __syncthreads();
    }
    float mean = r[0] * (1.0f / 256.0f);
    __syncthreads();
    float d = v - mean;
    r[tid] = d * d;
    __syncthreads();
    for (int o = 128; o > 0; o >>= 1) {
        if (tid < o) r[tid] += r[tid + o];
        __syncthreads();
    }
    float var = r[0] * (1.0f / 256.0f);
    out[idx] = d * rsqrtf(var + 1e-5f) * g[tid] + beta[tid];
}

// ---------------------------------------------------------------------------
// Depthwise causal conv (K=4) + SiLU over the x half of xz.
// ---------------------------------------------------------------------------
__global__ __launch_bounds__(256)
void conv_silu_kernel(const float* __restrict__ xz, const float* __restrict__ cw,
                      const float* __restrict__ cb, float* __restrict__ xc) {
    const int idx = blockIdx.x * 256 + threadIdx.x;   // over 8192*256
    const int d   = idx & 255;
    const int row = idx >> 8;
    const int t   = row & (SEQ_L - 1);
    float acc = cb[d];
    #pragma unroll
    for (int j = 0; j < 4; ++j) {
        int tt = t - 3 + j;
        if (tt >= 0) acc += cw[d * 4 + j] * xz[(size_t)(row - 3 + j) * 512 + d];
    }
    xc[(size_t)row * 256 + d] = silu_f(acc);
}

// ---------------------------------------------------------------------------
// Fused SSM scan: dt-proj + softplus, 16-state recurrence, C contraction,
// D skip, SiLU(z) gate.  grid = B (16), block = 256 (channel per thread).
// The (dt_in[4] | B[16] | C[16]) rows are staged into LDS in 32-step chunks
// via GLOBAL_LOAD_ASYNC_TO_LDS_B32, double-buffered so chunk c+1's DMA
// overlaps chunk c's recurrence. ASYNCcnt drained with s_wait_asynccnt.
// ---------------------------------------------------------------------------
#define CHUNK_T   32
#define CHUNK_F   (CHUNK_T * 36)     // 1152 floats per chunk
#define N_CHUNKS  (SEQ_L / CHUNK_T)  // 16

__global__ __launch_bounds__(256)
void ssm_scan_kernel(const float* __restrict__ xz, const float* __restrict__ xc,
                     const float* __restrict__ dbc,
                     const float* __restrict__ dtw, const float* __restrict__ dtb,
                     const float* __restrict__ alog, const float* __restrict__ dpar,
                     float* __restrict__ yg) {
    const int b = blockIdx.x;
    const int d = threadIdx.x;

    __shared__ float sd[2][CHUNK_F];

    const float w0 = dtw[0 * 256 + d];
    const float w1 = dtw[1 * 256 + d];
    const float w2 = dtw[2 * 256 + d];
    const float w3 = dtw[3 * 256 + d];
    const float dtbv = dtb[d];
    const float dpv  = dpar[d];

    float Ad[16];
    #pragma unroll
    for (int n = 0; n < 16; ++n) Ad[n] = -expf(alog[d * 16 + n]);

    float h[16];
    #pragma unroll
    for (int n = 0; n < 16; ++n) h[n] = 0.0f;

    const float* dbc_b = dbc + (size_t)b * SEQ_L * 36;

    // async-stage one chunk (uniform 5 issues/lane; clamped tail lanes
    // redundantly re-load the last word, keeping ASYNCcnt uniform per wave)
    auto issue_chunk = [&](int chunk, int buf) {
        unsigned lbase = (unsigned)(uintptr_t)(&sd[buf][0]);   // LDS byte offset
        const float* g = dbc_b + (size_t)chunk * CHUNK_F;
        #pragma unroll
        for (int i = 0; i < 5; ++i) {
            int s = d + i * 256;
            if (s > CHUNK_F - 1) s = CHUNK_F - 1;
            unsigned dst = lbase + (unsigned)s * 4u;
            unsigned long long src = (unsigned long long)(uintptr_t)(g + s);
            asm volatile("global_load_async_to_lds_b32 %0, %1, off"
                         :: "v"(dst), "v"(src) : "memory");
        }
    };

    issue_chunk(0, 0);

    for (int c = 0; c < N_CHUNKS; ++c) {
        const int buf = c & 1;
        asm volatile("s_wait_asynccnt 0x0" ::: "memory");
        __syncthreads();                       // chunk c visible to all waves;
                                               // also: everyone done reading c-1
        if (c + 1 < N_CHUNKS) issue_chunk(c + 1, buf ^ 1);

        for (int tt = 0; tt < CHUNK_T; ++tt) {
            const float* sr = &sd[buf][tt * 36];
            const size_t row = (size_t)b * SEQ_L + c * CHUNK_T + tt;

            float dt = softplus_f(w0 * sr[0] + w1 * sr[1] +
                                  w2 * sr[2] + w3 * sr[3] + dtbv);
            float xv = xc[row * 256 + d];
            float zv = xz[row * 512 + 256 + d];

            float y = 0.0f;
            #pragma unroll
            for (int n = 0; n < 16; ++n) {
                float da = expf(dt * Ad[n]);
                h[n] = da * h[n] + (dt * sr[4 + n]) * xv;
                y += h[n] * sr[20 + n];
            }
            yg[row * 256 + d] = (y + dpv * xv) * silu_f(zv);
        }
    }
}

// ---------------------------------------------------------------------------
// Host-side orchestration
// ---------------------------------------------------------------------------
template <int EPI, bool NGUARD, bool AREV, bool CREV>
static inline void gemm(hipStream_t s, const float* A, const float* B,
                        const float* bias, float* C,
                        int M, int N, int K, int lda, int ldb, int ldc) {
    dim3 grid((N + 127) / 128, M / 128);
    gemm_kernel<EPI, NGUARD, AREV, CREV><<<grid, 256, 0, s>>>(
        A, B, bias, C, M, N, K, lda, ldb, ldc);
}

extern "C" void kernel_launch(void* const* d_in, const int* in_sizes, int n_in,
                              void* d_out, int out_size, void* d_ws, size_t ws_size,
                              hipStream_t stream) {
    (void)in_sizes; (void)n_in; (void)out_size; (void)ws_size;

    const float* input_ids = (const float*)d_in[0];
    const float* proj_w    = (const float*)d_in[1];
    const float* proj_b    = (const float*)d_in[2];
    const float* ln0_g     = (const float*)d_in[3];
    const float* ln0_b     = (const float*)d_in[4];
    const float* ln1_g     = (const float*)d_in[5];
    const float* ln1_b     = (const float*)d_in[6];
    const float* ip_w      = (const float*)d_in[7];
    const float* ip_b      = (const float*)d_in[8];
    const float* s_inw     = (const float*)d_in[9];
    const float* s_convw   = (const float*)d_in[10];
    const float* s_convb   = (const float*)d_in[11];
    const float* s_xw      = (const float*)d_in[12];
    const float* s_dtw     = (const float*)d_in[13];
    const float* s_dtb     = (const float*)d_in[14];
    const float* s_alog    = (const float*)d_in[15];
    const float* s_d       = (const float*)d_in[16];
    const float* s_outw    = (const float*)d_in[17];
    const float* op_w      = (const float*)d_in[18];
    const float* op_b      = (const float*)d_in[19];
    const float* ln2_g     = (const float*)d_in[20];
    const float* ln2_b     = (const float*)d_in[21];
    const float* f_w1      = (const float*)d_in[22];
    const float* f_b1      = (const float*)d_in[23];
    const float* f_w2      = (const float*)d_in[24];
    const float* f_b2      = (const float*)d_in[25];
    const float* ln3_g     = (const float*)d_in[26];
    const float* ln3_b     = (const float*)d_in[27];

    const int NL = 12;
    const int M  = 16 * SEQ_L;           // 8192 rows
    const size_t M2 = (size_t)M * 256;

    // workspace carve-out (fp32)
    float* W    = (float*)d_ws;
    float* X    = W;                                  // residual stream
    float* X2   = X    + M2;
    float* XZ0  = X2   + M2;                          // (M,512)
    float* XZ1  = XZ0  + (size_t)M * 512;
    float* XC0  = XZ1  + (size_t)M * 512;
    float* XC1  = XC0  + M2;
    float* YG0  = XC1  + M2;
    float* YG1  = YG0  + M2;
    float* YS   = YG1  + M2;                          // concat of dir outs
    float* DBC0 = YS   + M2;                          // (M,36)
    float* DBC1 = DBC0 + (size_t)M * 36;
    float* LNX  = DBC1 + (size_t)M * 36;
    float* Hbuf = LNX  + M2;
    float* TMP  = Hbuf + M2;                          // (M,1024)
    float* TMP2 = TMP  + (size_t)M * 1024;            // (M,256)

    // ---- embed: x = LN0(input_ids @ proj_w + proj_b) ----
    gemm<1, false, false, false>(stream, input_ids, proj_w, proj_b, TMP,
                                 M, 256, 32, 32, 256, 256);
    ln_kernel<<<M, 256, 0, stream>>>(TMP, nullptr, ln0_g, ln0_b, X);

    for (int l = 0; l < NL; ++l) {
        // LN1 + input projection
        ln_kernel<<<M, 256, 0, stream>>>(X, nullptr, ln1_g + l * 256, ln1_b + l * 256, LNX);
        gemm<1, false, false, false>(stream, LNX, ip_w + (size_t)l * 256 * 256,
                                     ip_b + l * 256, Hbuf, M, 256, 256, 256, 256, 256);

        for (int dir = 0; dir < 2; ++dir) {
            const int ld = l * 2 + dir;
            float* XZ  = dir ? XZ1  : XZ0;
            float* XC  = dir ? XC1  : XC0;
            float* DBC = dir ? DBC1 : DBC0;
            float* YG  = dir ? YG1  : YG0;

            // xz = u @ inw  (backward branch: time-reversed A read)
            if (dir == 0)
                gemm<0, false, false, false>(stream, Hbuf, s_inw + (size_t)ld * 128 * 512,
                                             nullptr, XZ, M, 512, 128, 256, 512, 512);
            else
                gemm<0, false, true, false>(stream, Hbuf + 128, s_inw + (size_t)ld * 128 * 512,
                                            nullptr, XZ, M, 512, 128, 256, 512, 512);

            // depthwise conv + silu
            conv_silu_kernel<<<M, 256, 0, stream>>>(
                XZ, s_convw + (size_t)ld * 256 * 4, s_convb + (size_t)ld * 256, XC);

            // dbc = x @ xw   (N=36 -> guarded variant)
            gemm<0, true, false, false>(stream, XC, s_xw + (size_t)ld * 256 * 36,
                                        nullptr, DBC, M, 36, 256, 256, 36, 36);

            // fused dt / scan / gate (async LDS staging)
            ssm_scan_kernel<<<16, 256, 0, stream>>>(
                XZ, XC, DBC,
                s_dtw  + (size_t)ld * 4 * 256, s_dtb + (size_t)ld * 256,
                s_alog + (size_t)ld * 256 * 16, s_d  + (size_t)ld * 256, YG);

            // y @ outw -> concat slot (backward branch: time-reversed C write)
            if (dir == 0)
                gemm<0, false, false, false>(stream, YG, s_outw + (size_t)ld * 256 * 128,
                                             nullptr, YS, M, 128, 256, 256, 128, 256);
            else
                gemm<0, false, false, true>(stream, YG, s_outw + (size_t)ld * 256 * 128,
                                            nullptr, YS + 128, M, 128, 256, 256, 128, 256);
        }

        // out-proj + residual + LN2
        gemm<1, false, false, false>(stream, YS, op_w + (size_t)l * 256 * 256,
                                     op_b + l * 256, TMP, M, 256, 256, 256, 256, 256);
        ln_kernel<<<M, 256, 0, stream>>>(TMP, X, ln2_g + l * 256, ln2_b + l * 256, X2);

        // FFN: gelu(X2 @ w1 + b1) @ w2 + b2, residual + LN3
        gemm<2, false, false, false>(stream, X2, f_w1 + (size_t)l * 256 * 1024,
                                     f_b1 + l * 1024, TMP, M, 1024, 256, 256, 1024, 1024);
        gemm<1, false, false, false>(stream, TMP, f_w2 + (size_t)l * 1024 * 256,
                                     f_b2 + l * 256, TMP2, M, 256, 1024, 1024, 256, 256);

        float* dst = (l == NL - 1) ? (float*)d_out : X;
        ln_kernel<<<M, 256, 0, stream>>>(TMP2, X2, ln3_g + l * 256, ln3_b + l * 256, dst);
    }
}
